// ExplaiNN3_10788957848060
// MI455X (gfx1250) — compile-verified, matrix operand order checked by reading, not run on hardware
//
#include <hip/hip_runtime.h>
#include <hip/hip_bf16.h>

typedef __attribute__((ext_vector_type(16))) __bf16 v16bf;
typedef __attribute__((ext_vector_type(8)))  float  v8f;
typedef __attribute__((ext_vector_type(4)))  unsigned v4u;

#define U_CNT   300
#define BATCH   256
#define INLEN   608
#define K1      76     // true K = 4*19
#define K1R     80     // reordered K = 4*20 (tap 19 zero-weighted)
#define K1P     96
#define LPOOL   84
#define LPOOLP  96     // padded K for stage-2 GEMM
#define HID     100
#define NCLS    50
#define EPS_F   1e-5f
#define XROW    640    // padded x row (zeros past 608); row 4 = all zero
#define XTOT    (5 * XROW)
#define ZSTR    320    // zbuf row stride (zero-padded K for final GEMM)

union Frag {
    v16bf v;
    v4u q[2];
    unsigned u[8];
    __bf16 e[16];
};

__device__ __forceinline__ __bf16 f2bf(float f) {
    unsigned u = __builtin_bit_cast(unsigned, f);
    u += 0x7FFFu + ((u >> 16) & 1u);           // round-to-nearest-even
    unsigned short h = (unsigned short)(u >> 16);
    return __builtin_bit_cast(__bf16, h);
}

__device__ __forceinline__ v8f wmma_bf16(const Frag& a, const Frag& b, v8f c) {
    return __builtin_amdgcn_wmma_f32_16x16x32_bf16(
        false, a.v, false, b.v, (short)0, c, false, false);
}

// ---------------------------------------------------------------------------
// Kernel 1: implicit-GEMM conv(4->300,k=19) + bias + BN + exp + maxpool7
// grid (19, 256): blockIdx.x = unit tile (16 units), blockIdx.y = batch
// block = 128 threads (4 waves). Output: y_pool[u][b][96] bf16 (K zero-padded).
// K ordering = c*20+t so every WMMA bf16 pair is a contiguous u32 in the x
// image; a second, one-element-shifted copy of the image makes every pair
// 4B-aligned for whichever (t+n) parity a lane has (lane-constant choice).
// ---------------------------------------------------------------------------
__global__ __launch_bounds__(128)
void k1_conv_bn_exp_pool(const float* __restrict__ x,
                         const float* __restrict__ w1,
                         const float* __restrict__ b1,
                         const float* __restrict__ g1,
                         const float* __restrict__ be1,
                         const float* __restrict__ m1,
                         const float* __restrict__ v1,
                         __bf16* __restrict__ y_pool) {
    __shared__ __bf16 xall[2 * XTOT];       // [0..XTOT): image, [XTOT..): image shifted +1
    __shared__ __bf16 af[16 * 104];         // A operand, 16B-aligned conflict-free rows
    __shared__ float  alpha_s[16];
    __shared__ float  beta_s[16];
    __shared__ float  eb[16 * 592];         // exp(conv) staging for pooling

    const int u0   = blockIdx.x * 16;
    const int b    = blockIdx.y;
    const int tid  = threadIdx.x;
    const int wid  = tid >> 5;
    const int lane = tid & 31;

    // ---- stage x[b] as bf16, twice (second copy shifted by one element) ----
    const float* xrow = x + (size_t)b * (4 * INLEN);
    const __bf16 bz = f2bf(0.0f);
    for (int i = tid; i < 2 * XTOT; i += 128) {
        int e = (i >= XTOT) ? (i - XTOT - 1) : i;   // logical element (-1 => zero)
        __bf16 v = bz;
        if (e >= 0) {
            int row = e / XROW, col = e - row * XROW;
            if (row < 4 && col < INLEN) v = f2bf(xrow[row * INLEN + col]);
        }
        xall[i] = v;
    }
    // ---- stage conv weights in k = c*20+t order (t==19 column zeroed) ----
    for (int i = tid; i < 16 * K1P; i += 128) {
        int m = i / K1P, k = i - m * K1P;
        int u = u0 + m;
        float w = 0.0f;
        if (u < U_CNT && k < K1R) {
            int c = k / 20, t = k - c * 20;
            if (t < 19) w = w1[(size_t)u * K1 + c * 19 + t];
        }
        af[m * 104 + k] = f2bf(w);
    }
    // ---- fold bias+BN: y*alpha + beta ----
    if (tid < 16) {
        int u = u0 + tid;
        if (u < U_CNT) {
            float a = g1[u] * rsqrtf(v1[u] + EPS_F);
            alpha_s[tid] = a;
            beta_s[tid]  = (b1[u] - m1[u]) * a + be1[u];
        } else { alpha_s[tid] = 0.0f; beta_s[tid] = 0.0f; }
    }
    __syncthreads();

    const int n     = lane & 15;
    const int khalf = lane >> 4;

    // ---- load the 3 A fragments once per wave (b128 LDS loads) ----
    Frag a[3];
#pragma unroll
    for (int ks = 0; ks < 3; ++ks) {
        int k0 = ks * 32 + khalf * 8;
        a[ks].q[0] = *(const v4u*)&af[n * 104 + k0];
        a[ks].q[1] = *(const v4u*)&af[n * 104 + k0 + 16];
    }

    // ---- precompute the 24 aligned-u32 gather offsets (lane constants) ----
    int eadr[3][8];
#pragma unroll
    for (int ks = 0; ks < 3; ++ks)
#pragma unroll
        for (int j = 0; j < 8; ++j) {
            int kb = ks * 32 + ((j < 4) ? 0 : 16) + khalf * 8 + 2 * (j & 3);
            int idx;
            if (kb < K1R) {
                int c = kb / 20, t = kb - c * 20;
                idx = c * XROW + t + n;
            } else {
                idx = 4 * XROW + n;             // zero row
            }
            eadr[ks][j] = (idx & 1) ? (XTOT + idx + 1) : idx;  // pick aligned copy
        }

    // ---- 37 position tiles of 16, striped over 4 waves ----
    const int mbase = (lane < 16) ? 0 : 8;
    for (int t = wid; t < 37; t += 4) {
        const int l0 = t * 16;
        v8f acc;
#pragma unroll
        for (int r = 0; r < 8; ++r) acc[r] = 0.0f;

#pragma unroll
        for (int ks = 0; ks < 3; ++ks) {
            Frag bf_;
#pragma unroll
            for (int j = 0; j < 8; ++j)
                bf_.u[j] = *(const unsigned*)&xall[eadr[ks][j] + l0];
            acc = wmma_bf16(a[ks], bf_, acc);
        }

        // epilogue: BN + exp, into LDS pooling buffer
#pragma unroll
        for (int r = 0; r < 8; ++r) {
            int m = mbase + r;
            float val = acc[r] * alpha_s[m] + beta_s[m];
            eb[m * 592 + l0 + n] = __expf(val);
        }
    }
    __syncthreads();

    // ---- maxpool(7,7): 590 -> 84 (+ zero pad to 96), y_pool[u][b][96] bf16 ----
    for (int idx = tid; idx < 16 * LPOOLP; idx += 128) {
        int m = idx / LPOOLP, f = idx - m * LPOOLP;
        int u = u0 + m;
        if (u < U_CNT) {
            float mx = 0.0f;
            if (f < LPOOL) {
                const float* row = &eb[m * 592 + f * 7];
                mx = row[0];
#pragma unroll
                for (int q = 1; q < 7; ++q) mx = fmaxf(mx, row[q]);
            }
            y_pool[((size_t)u * BATCH + b) * LPOOLP + f] = f2bf(mx);
        }
    }
}

// ---------------------------------------------------------------------------
// Kernel 2: per-unit MLP 84->100 (+BN,ReLU) fused with 100->1 (+BN,ReLU)
// grid = 320 (one unit per block; blocks 300..319 zero-fill zbuf padding)
// 256 threads (8 waves, 2 concurrent M-tiles each). zbuf[b][u] bf16, stride 320.
// ---------------------------------------------------------------------------
__global__ __launch_bounds__(256)
void k2_unit_mlp(const __bf16* __restrict__ y_pool,
                 const float* __restrict__ w2,  const float* __restrict__ b2,
                 const float* __restrict__ g2,  const float* __restrict__ be2,
                 const float* __restrict__ m2,  const float* __restrict__ v2,
                 const float* __restrict__ w3,  const float* __restrict__ b3,
                 const float* __restrict__ g3,  const float* __restrict__ be3,
                 const float* __restrict__ m3,  const float* __restrict__ v3,
                 __bf16* __restrict__ zbuf) {
    __shared__ __bf16 btT[112 * 104];   // w2[u]^T padded: [o][k], 16B-aligned rows
    __shared__ float  A2l[112], B2l[112], w3l[112];

    const int u    = blockIdx.x;
    const int tid  = threadIdx.x;
    const int wid  = tid >> 5;
    const int lane = tid & 31;

    if (u >= U_CNT) {                   // zero-fill zbuf pad columns for kernel 3
        for (int bb = tid; bb < BATCH; bb += 256)
            zbuf[(size_t)bb * ZSTR + u] = f2bf(0.0f);
        return;
    }

    __builtin_prefetch(y_pool + (size_t)u * BATCH * LPOOLP, 0, 3);

    // stage transposed/padded w2[u] as bf16
    const __bf16 bz = f2bf(0.0f);
    for (int idx = tid; idx < 112 * K1P; idx += 256) {
        int o = idx / K1P, k = idx - o * K1P;
        btT[o * 104 + k] = (o < HID && k < LPOOL)
            ? f2bf(w2[(size_t)u * HID * LPOOL + o * LPOOL + k]) : bz;
    }
    // fold BN2 into scale/shift; zero-pad so o>=100 contributes nothing
    for (int o = tid; o < 112; o += 256) {
        if (o < HID) {
            size_t i = (size_t)u * HID + o;
            float a = g2[i] * rsqrtf(v2[i] + EPS_F);
            A2l[o] = a;
            B2l[o] = (b2[i] - m2[i]) * a + be2[i];
            w3l[o] = w3[i];
        } else { A2l[o] = 0.0f; B2l[o] = 0.0f; w3l[o] = 0.0f; }
    }
    __syncthreads();

    const float al3  = g3[u] * rsqrtf(v3[u] + EPS_F);
    const float m3v  = m3[u], be3v = be3[u], b3v = b3[u];

    const int n     = lane & 15;
    const int khalf = lane >> 4;
    const int mbase = (lane < 16) ? 0 : 8;
    const int b0    = wid * 32;         // two 16-row M-tiles per wave

    // A fragments straight from global y_pool (b128 loads, rows zero-padded to 96)
    Frag a[2][3];
#pragma unroll
    for (int mt2 = 0; mt2 < 2; ++mt2) {
        const __bf16* arow = y_pool + ((size_t)u * BATCH + b0 + mt2 * 16 + n) * LPOOLP;
#pragma unroll
        for (int ks = 0; ks < 3; ++ks) {
            int k0 = ks * 32 + khalf * 8;
            a[mt2][ks].q[0] = *(const v4u*)(arow + k0);
            a[mt2][ks].q[1] = *(const v4u*)(arow + k0 + 16);
        }
    }

    v8f acc[2][7];
#pragma unroll
    for (int mt2 = 0; mt2 < 2; ++mt2)
#pragma unroll
        for (int nt = 0; nt < 7; ++nt)
#pragma unroll
            for (int r = 0; r < 8; ++r) acc[mt2][nt][r] = 0.0f;

#pragma unroll
    for (int ks = 0; ks < 3; ++ks) {
        int k0 = ks * 32 + khalf * 8;
#pragma unroll
        for (int nt = 0; nt < 7; ++nt) {
            Frag bf_;
            bf_.q[0] = *(const v4u*)&btT[(nt * 16 + n) * 104 + k0];
            bf_.q[1] = *(const v4u*)&btT[(nt * 16 + n) * 104 + k0 + 16];
            acc[0][nt] = wmma_bf16(a[0][ks], bf_, acc[0][nt]);
            acc[1][nt] = wmma_bf16(a[1][ks], bf_, acc[1][nt]);
        }
    }

    // fused epilogue per M-tile: BN2 + ReLU + dot with w3, cross-lane reduce
#pragma unroll
    for (int mt2 = 0; mt2 < 2; ++mt2) {
        float part[8];
#pragma unroll
        for (int r = 0; r < 8; ++r) part[r] = 0.0f;
#pragma unroll
        for (int nt = 0; nt < 7; ++nt) {
            int o = nt * 16 + n;
            float sc = A2l[o], sh = B2l[o], wv = w3l[o];
#pragma unroll
            for (int r = 0; r < 8; ++r) {
                float h = fmaxf(acc[mt2][nt][r] * sc + sh, 0.0f);
                part[r] += h * wv;
            }
        }
#pragma unroll
        for (int r = 0; r < 8; ++r) {
            part[r] += __shfl_xor(part[r], 1, 32);
            part[r] += __shfl_xor(part[r], 2, 32);
            part[r] += __shfl_xor(part[r], 4, 32);
            part[r] += __shfl_xor(part[r], 8, 32);
        }
        if ((lane & 15) == 0) {
#pragma unroll
            for (int r = 0; r < 8; ++r) {
                int bb = b0 + mt2 * 16 + mbase + r;
                float z = part[r] + b3v;
                z = (z - m3v) * al3 + be3v;
                z = fmaxf(z, 0.0f);
                zbuf[(size_t)bb * ZSTR + u] = f2bf(z);
            }
        }
    }
}

// ---------------------------------------------------------------------------
// Kernel 3: final linear [256 x 300] @ wf^T + bf -> [256 x 50] f32
// grid = 2 blocks, 256 threads (8 waves, one 16-batch tile each)
// ---------------------------------------------------------------------------
__global__ __launch_bounds__(256)
void k3_final(const __bf16* __restrict__ zbuf,
              const float* __restrict__ wf,
              const float* __restrict__ bfv,
              float* __restrict__ out) {
    __shared__ __bf16 wT[64 * 328];   // wf^T padded: [n][k], 16B-aligned rows
    __shared__ float  bfl[64];

    const int tid  = threadIdx.x;
    const int wid  = tid >> 5;
    const int lane = tid & 31;

    const __bf16 bz = f2bf(0.0f);
    for (int idx = tid; idx < 64 * ZSTR; idx += 256) {
        int nn = idx / ZSTR, k = idx - nn * ZSTR;
        wT[nn * 328 + k] = (nn < NCLS && k < U_CNT) ? f2bf(wf[(size_t)nn * U_CNT + k]) : bz;
    }
    if (tid < 64) bfl[tid] = (tid < NCLS) ? bfv[tid] : 0.0f;
    __syncthreads();

    const int mt    = blockIdx.x * 8 + wid;
    const int n     = lane & 15;
    const int khalf = lane >> 4;
    const int mbase = (lane < 16) ? 0 : 8;

    const __bf16* zrow = zbuf + (size_t)(mt * 16 + n) * ZSTR;

    v8f acc[4];
#pragma unroll
    for (int nt = 0; nt < 4; ++nt)
#pragma unroll
        for (int r = 0; r < 8; ++r) acc[nt][r] = 0.0f;

    for (int ks = 0; ks < 10; ++ks) {
        int k0 = ks * 32 + khalf * 8;
        Frag a;
        a.q[0] = *(const v4u*)(zrow + k0);
        a.q[1] = *(const v4u*)(zrow + k0 + 16);
#pragma unroll
        for (int nt = 0; nt < 4; ++nt) {
            Frag bb;
            bb.q[0] = *(const v4u*)&wT[(nt * 16 + n) * 328 + k0];
            bb.q[1] = *(const v4u*)&wT[(nt * 16 + n) * 328 + k0 + 16];
            acc[nt] = wmma_bf16(a, bb, acc[nt]);
        }
    }

#pragma unroll
    for (int nt = 0; nt < 4; ++nt) {
        int o = nt * 16 + n;
        if (o < NCLS) {
#pragma unroll
            for (int r = 0; r < 8; ++r) {
                int b = mt * 16 + mbase + r;
                out[(size_t)b * NCLS + o] = acc[nt][r] + bfl[o];
            }
        }
    }
}

// ---------------------------------------------------------------------------
extern "C" void kernel_launch(void* const* d_in, const int* in_sizes, int n_in,
                              void* d_out, int out_size, void* d_ws, size_t ws_size,
                              hipStream_t stream) {
    const float* x   = (const float*)d_in[0];
    const float* w1  = (const float*)d_in[1];
    const float* b1  = (const float*)d_in[2];
    const float* g1  = (const float*)d_in[3];
    const float* be1 = (const float*)d_in[4];
    const float* m1  = (const float*)d_in[5];
    const float* v1  = (const float*)d_in[6];
    const float* w2  = (const float*)d_in[7];
    const float* b2  = (const float*)d_in[8];
    const float* g2  = (const float*)d_in[9];
    const float* be2 = (const float*)d_in[10];
    const float* m2  = (const float*)d_in[11];
    const float* v2  = (const float*)d_in[12];
    const float* w3  = (const float*)d_in[13];
    const float* b3  = (const float*)d_in[14];
    const float* g3  = (const float*)d_in[15];
    const float* be3 = (const float*)d_in[16];
    const float* m3  = (const float*)d_in[17];
    const float* v3  = (const float*)d_in[18];
    const float* wf  = (const float*)d_in[19];
    const float* bf  = (const float*)d_in[20];
    float* out = (float*)d_out;

    // workspace layout
    __bf16* y_pool = (__bf16*)d_ws;   // 300*256*96 bf16 = 14,745,600 B
    __bf16* zbuf   = (__bf16*)((char*)d_ws + (size_t)U_CNT * BATCH * LPOOLP * 2); // 256*320 bf16

    dim3 g1d(19, BATCH);
    k1_conv_bn_exp_pool<<<g1d, 128, 0, stream>>>(x, w1, b1, g1, be1, m1, v1, y_pool);

    k2_unit_mlp<<<ZSTR, 256, 0, stream>>>(y_pool, w2, b2, g2, be2, m2, v2,
                                          w3, b3, g3, be3, m3, v3, zbuf);

    k3_final<<<2, 256, 0, stream>>>(zbuf, wf, bf, out);
}
